// BaselineTransformerLayer_89687507076175
// MI455X (gfx1250) — matrix operand order repaired and checked by reading, not compiled, not run on hardware
//
#include <hip/hip_runtime.h>
#include <hip/hip_bf16.h>

// ---------------------------------------------------------------------------
// Types for CDNA5 WMMA
// ---------------------------------------------------------------------------
typedef __bf16 bf16_t;
typedef __attribute__((ext_vector_type(16))) __bf16 v16bf;
typedef __attribute__((ext_vector_type(8)))  __bf16 v8bf;
typedef __attribute__((ext_vector_type(4)))  __bf16 v4bf;
typedef __attribute__((ext_vector_type(8)))  float  v8f;

#define TM 128
#define TN 128
#define TK 64
#define LDSK (TK + 8)   // 72 bf16 = 144B row stride, rows stay 16B aligned

// ---------------------------------------------------------------------------
// Generic GEMM:  C[M,N] = alpha * A[M,K] * op(B)
//   B_KN == 0 : B is [N][K] row-major (computes A @ B^T)   ("TN")
//   B_KN == 1 : B is [K][N] row-major (computes A @ B)     ("NN")
// fp32 in/out, bf16 WMMA compute (downconvert during LDS staging).
// Requirements honored by every launch below:
//   K % 64 == 0, M % 128 == 0, N % 4 == 0, leading dims % 4 == 0,
//   base pointers 16B aligned. Columns >= N use clamped (legal) addresses
//   and are discarded at the store.
// causal != 0 : skip output blocks entirely above the diagonal.
// ---------------------------------------------------------------------------
template <int B_KN>
__global__ __launch_bounds__(256)
void gemm_bf16_wmma(const float* __restrict__ A, const float* __restrict__ B,
                    float* __restrict__ C, int M, int N, int K,
                    int lda, int ldb, int ldc, float alpha, int causal)
{
    __shared__ __align__(16) bf16_t As[TM][LDSK];
    __shared__ __align__(16) bf16_t Bs[TN][LDSK];

    const int bm = blockIdx.y, bn = blockIdx.x;
    const int m_base = bm * TM, n_base = bn * TN;
    if (causal && n_base > m_base + TM - 1) return;   // fully-masked block

    const int tid  = threadIdx.x;
    const int lane = tid & 31;
    const int wave = tid >> 5;
    const int wr   = wave & 3;     // wave row (M): 0..3, 32 rows each
    const int wc   = wave >> 2;    // wave col (N): 0..1, 64 cols each
    const int half = lane >> 4;    // 0: lanes 0-15, 1: lanes 16-31
    const int l16  = lane & 15;

    v8f acc[2][4];
#pragma unroll
    for (int i = 0; i < 2; ++i)
#pragma unroll
        for (int j = 0; j < 4; ++j) { v8f z = {}; acc[i][j] = z; }

    float4 aR[8], bR[8];

    // ------------------------------------------------------------------
    // staging helpers (branch-free, clamped addresses)
    // A tile: 128 rows x 64 cols f32 -> idx -> (row = idx/16, c = idx%16*4)
    // ------------------------------------------------------------------
    auto load_A = [&](int k0) {
#pragma unroll
        for (int e = 0; e < 8; ++e) {
            int idx = e * 256 + tid;
            int row = idx >> 4;            // 0..127
            int c   = (idx & 15) * 4;      // 0,4,..,60
            int gm  = m_base + row;        // M % 128 == 0 -> in range
            aR[e] = *(const float4*)(A + (long)gm * lda + k0 + c);
        }
    };
    auto load_B = [&](int k0) {
        if (B_KN) {
#pragma unroll
            for (int e = 0; e < 8; ++e) {
                int idx = e * 256 + tid;
                int kk  = idx >> 5;            // 0..63
                int n   = (idx & 31) * 4;      // 0,4,..,124
                int gn  = n_base + n;
                gn = gn <= N - 4 ? gn : N - 4; // clamp, stays legal
                bR[e] = *(const float4*)(B + (long)(k0 + kk) * ldb + gn);
            }
        } else {
#pragma unroll
            for (int e = 0; e < 8; ++e) {
                int idx = e * 256 + tid;
                int n   = idx >> 4;            // 0..127
                int c   = (idx & 15) * 4;
                int gn  = n_base + n;
                gn = gn <= N - 1 ? gn : N - 1; // clamp row
                bR[e] = *(const float4*)(B + (long)gn * ldb + k0 + c);
            }
        }
    };
    auto store_A = [&]() {
#pragma unroll
        for (int e = 0; e < 8; ++e) {
            int idx = e * 256 + tid;
            int row = idx >> 4;
            int c   = (idx & 15) * 4;
            v4bf v; v[0] = (bf16_t)aR[e].x; v[1] = (bf16_t)aR[e].y;
                    v[2] = (bf16_t)aR[e].z; v[3] = (bf16_t)aR[e].w;
            *(v4bf*)&As[row][c] = v;
        }
    };
    auto store_B = [&]() {
        if (B_KN) {
#pragma unroll
            for (int e = 0; e < 8; ++e) {
                int idx = e * 256 + tid;
                int kk  = idx >> 5;
                int n   = (idx & 31) * 4;
                Bs[n + 0][kk] = (bf16_t)bR[e].x;
                Bs[n + 1][kk] = (bf16_t)bR[e].y;
                Bs[n + 2][kk] = (bf16_t)bR[e].z;
                Bs[n + 3][kk] = (bf16_t)bR[e].w;
            }
        } else {
#pragma unroll
            for (int e = 0; e < 8; ++e) {
                int idx = e * 256 + tid;
                int n   = idx >> 4;
                int c   = (idx & 15) * 4;
                v4bf v; v[0] = (bf16_t)bR[e].x; v[1] = (bf16_t)bR[e].y;
                        v[2] = (bf16_t)bR[e].z; v[3] = (bf16_t)bR[e].w;
                *(v4bf*)&Bs[n][c] = v;
            }
        }
    };

    // ---- prologue: fetch first K-tile into registers ----
    load_A(0);
    load_B(0);

    for (int k0 = 0; k0 < K; k0 += TK) {
        store_A();
        store_B();
        __syncthreads();

        // issue next tile's global loads early so they overlap the wmma work
        if (k0 + TK < K) { load_A(k0 + TK); load_B(k0 + TK); }

        // ---- two K=32 sub-steps per staged tile: 16 wmma per barrier ----
#pragma unroll
        for (int ks = 0; ks < TK; ks += 32) {
            // A 16x32 bf16: lane holds M=l16; elems 0-7 -> K = ks+half*8+e,
            //               elems 8-15 -> K = ks+16+half*8+(e-8)
            v16bf afrag[2];
#pragma unroll
            for (int mt = 0; mt < 2; ++mt) {
                const bf16_t* ar = &As[wr * 32 + mt * 16 + l16][ks];
                v8bf lo = *(const v8bf*)(ar + half * 8);
                v8bf hi = *(const v8bf*)(ar + 16 + half * 8);
                v16bf a;
#pragma unroll
                for (int e = 0; e < 8; ++e) { a[e] = lo[e]; a[e + 8] = hi[e]; }
                afrag[mt] = a;
            }
            // B 32x16 bf16: lane holds N=l16; elems 0-7 -> K = ks+half*8+e,
            //               elems 8-15 -> K = ks+16+half*8+(e-8)
            v16bf bfrag[4];
#pragma unroll
            for (int nt = 0; nt < 4; ++nt) {
                const bf16_t* br = &Bs[wc * 64 + nt * 16 + l16][ks];
                v8bf blo = *(const v8bf*)(br + half * 8);
                v8bf bhi = *(const v8bf*)(br + 16 + half * 8);
                v16bf b;
#pragma unroll
                for (int e = 0; e < 8; ++e) { b[e] = blo[e]; b[e + 8] = bhi[e]; }
                bfrag[nt] = b;
            }

#pragma unroll
            for (int mt = 0; mt < 2; ++mt)
#pragma unroll
                for (int nt = 0; nt < 4; ++nt)
                    acc[mt][nt] = __builtin_amdgcn_wmma_f32_16x16x32_bf16(
                        false, afrag[mt], false, bfrag[nt],
                        (short)0, acc[mt][nt], false, false);
        }
        __syncthreads();
    }

    // ---- store C: f32 16x16 layout: VGPR r -> row = base + half*8 + r, col = l16
#pragma unroll
    for (int mt = 0; mt < 2; ++mt)
#pragma unroll
        for (int nt = 0; nt < 4; ++nt) {
            int col = n_base + wc * 64 + nt * 16 + l16;
            if (col >= N) continue;
#pragma unroll
            for (int r = 0; r < 8; ++r) {
                int row = m_base + wr * 32 + mt * 16 + half * 8 + r;
                C[(long)row * ldc + col] = alpha * acc[mt][nt][r];
            }
        }
}

// ---------------------------------------------------------------------------
// LayerNorm (one block per row)
// ---------------------------------------------------------------------------
__global__ __launch_bounds__(256)
void layernorm_kernel(const float* __restrict__ x, const float* __restrict__ w,
                      const float* __restrict__ b, float* __restrict__ y, int H)
{
    long row = blockIdx.x;
    const float* xr = x + row * H;
    float* yr = y + row * H;
    __shared__ float sred[8];
    __shared__ float smean, svar;

    float s = 0.f;
    for (int i = threadIdx.x; i < H; i += 256) s += xr[i];
    for (int o = 16; o > 0; o >>= 1) s += __shfl_xor(s, o, 32);
    if ((threadIdx.x & 31) == 0) sred[threadIdx.x >> 5] = s;
    __syncthreads();
    if (threadIdx.x == 0) { float t = 0; for (int i = 0; i < 8; ++i) t += sred[i]; smean = t / H; }
    __syncthreads();
    float mean = smean;

    float v = 0.f;
    for (int i = threadIdx.x; i < H; i += 256) { float d = xr[i] - mean; v += d * d; }
    for (int o = 16; o > 0; o >>= 1) v += __shfl_xor(v, o, 32);
    if ((threadIdx.x & 31) == 0) sred[threadIdx.x >> 5] = v;
    __syncthreads();
    if (threadIdx.x == 0) { float t = 0; for (int i = 0; i < 8; ++i) t += sred[i]; svar = t / H; }
    __syncthreads();

    float inv = rsqrtf(svar + 1e-5f);
    for (int i = threadIdx.x; i < H; i += 256)
        yr[i] = (xr[i] - mean) * inv * w[i] + b[i];
}

// ---------------------------------------------------------------------------
// Causal softmax over one row of an SxS score matrix (in place, f32)
// ---------------------------------------------------------------------------
__global__ __launch_bounds__(256)
void softmax_causal_kernel(float* __restrict__ sc, int S)
{
    long row = blockIdx.x;
    float* sr = sc + row * S;
    int n = (int)row + 1;          // valid columns: 0..row
    __shared__ float sred[8];
    __shared__ float sbc;

    float mx = -3.0e38f;
    for (int i = threadIdx.x; i < n; i += 256) mx = fmaxf(mx, sr[i]);
    for (int o = 16; o > 0; o >>= 1) mx = fmaxf(mx, __shfl_xor(mx, o, 32));
    if ((threadIdx.x & 31) == 0) sred[threadIdx.x >> 5] = mx;
    __syncthreads();
    if (threadIdx.x == 0) { float t = sred[0]; for (int i = 1; i < 8; ++i) t = fmaxf(t, sred[i]); sbc = t; }
    __syncthreads();
    float rmax = sbc;
    __syncthreads();

    float sum = 0.f;
    for (int i = threadIdx.x; i < n; i += 256) { float e = expf(sr[i] - rmax); sr[i] = e; sum += e; }
    for (int o = 16; o > 0; o >>= 1) sum += __shfl_xor(sum, o, 32);
    if ((threadIdx.x & 31) == 0) sred[threadIdx.x >> 5] = sum;
    __syncthreads();
    if (threadIdx.x == 0) { float t = 0; for (int i = 0; i < 8; ++i) t += sred[i]; sbc = t; }
    __syncthreads();
    float inv = 1.0f / sbc;

    for (int i = threadIdx.x; i < S; i += 256)
        sr[i] = (i < n) ? sr[i] * inv : 0.0f;
}

// ---------------------------------------------------------------------------
// Router: logits = x @ W (W is [H][E]), softmax, top-2, renorm -> gates[T][E]
// ---------------------------------------------------------------------------
__global__ __launch_bounds__(256)
void router_kernel(const float* __restrict__ x, const float* __restrict__ W,
                   float* __restrict__ gates, int H, int E)
{
    long t = blockIdx.x;
    int wave = threadIdx.x >> 5, lane = threadIdx.x & 31;
    __shared__ float logits[8];

    if (wave < E) {
        float s = 0.f;
        for (int h = lane; h < H; h += 32) s += x[t * H + h] * W[(long)h * E + wave];
        for (int o = 16; o > 0; o >>= 1) s += __shfl_xor(s, o, 32);
        if (lane == 0) logits[wave] = s;
    }
    __syncthreads();
    if (threadIdx.x == 0) {
        float mx = logits[0];
        for (int e = 1; e < E; ++e) mx = fmaxf(mx, logits[e]);
        float p[8], sum = 0.f;
        for (int e = 0; e < E; ++e) { p[e] = expf(logits[e] - mx); sum += p[e]; }
        for (int e = 0; e < E; ++e) p[e] /= sum;
        int i1 = 0;
        for (int e = 1; e < E; ++e) if (p[e] > p[i1]) i1 = e;
        int i2 = (i1 == 0) ? 1 : 0;
        for (int e = 0; e < E; ++e) { if (e == i1) continue; if (p[e] > p[i2]) i2 = e; }
        float gs = p[i1] + p[i2];
        for (int e = 0; e < E; ++e) gates[t * E + e] = 0.0f;
        gates[t * E + i1] = p[i1] / gs;
        gates[t * E + i2] = p[i2] / gs;
    }
}

// ---------------------------------------------------------------------------
// Elementwise helpers
// ---------------------------------------------------------------------------
__global__ __launch_bounds__(256)
void gelu_kernel(float* __restrict__ x, long n)
{
    long i = (long)blockIdx.x * 256 + threadIdx.x;
    long stride = (long)gridDim.x * 256;
    for (; i < n; i += stride) {
        float v = x[i];
        x[i] = 0.5f * v * (1.0f + erff(v * 0.70710678118654752f));
    }
}

__global__ __launch_bounds__(256)
void add_kernel(const float* __restrict__ a, const float* __restrict__ b,
                float* __restrict__ c, long n)
{
    long i = (long)blockIdx.x * 256 + threadIdx.x;
    long stride = (long)gridDim.x * 256;
    for (; i < n; i += stride) c[i] = a[i] + b[i];
}

__global__ __launch_bounds__(256)
void zero_kernel(float* __restrict__ a, long n)
{
    long i = (long)blockIdx.x * 256 + threadIdx.x;
    long stride = (long)gridDim.x * 256;
    for (; i < n; i += stride) a[i] = 0.0f;
}

// moe_acc[t, :] += gates[t*E + e] * y[t, :]
__global__ __launch_bounds__(256)
void scale_accum_kernel(const float* __restrict__ y, const float* __restrict__ gates,
                        float* __restrict__ acc, int H, int E, int e, long n)
{
    long i = (long)blockIdx.x * 256 + threadIdx.x;
    long stride = (long)gridDim.x * 256;
    for (; i < n; i += stride) {
        long t = i / H;
        float g = gates[t * E + e];
        if (g != 0.0f) acc[i] += g * y[i];
    }
}

// ---------------------------------------------------------------------------
// Host-side orchestration
// ---------------------------------------------------------------------------
extern "C" void kernel_launch(void* const* d_in, const int* in_sizes, int n_in,
                              void* d_out, int out_size, void* d_ws, size_t ws_size,
                              hipStream_t stream)
{
    const int H = 1024, QKVO = 1536, E = 8, F = 4096, D = 64, Hq = 16;
    const int S = in_sizes[0] / H;      // 2048 (B == 1)
    const long SH = (long)S * H;

    const float* hidden  = (const float*)d_in[0];
    const float* ln1_w   = (const float*)d_in[1];
    const float* ln1_b   = (const float*)d_in[2];
    const float* ln2_w   = (const float*)d_in[3];
    const float* ln2_b   = (const float*)d_in[4];
    const float* qkv_w   = (const float*)d_in[5];   // [1536][1024]
    const float* proj_w  = (const float*)d_in[6];   // [1024][1024]
    const float* rtr_w   = (const float*)d_in[7];   // [1024][8]
    const float* moe_w1  = (const float*)d_in[8];   // [8][1024][4096]
    const float* moe_w2  = (const float*)d_in[9];   // [8][4096][1024]
    float* out = (float*)d_out;

    // workspace layout (fp32)
    char* w = (char*)d_ws;
    float* ln1   = (float*)w;                           w += SH * 4;
    float* qkv   = (float*)w;                           w += (long)S * QKVO * 4;
    float* sc    = (float*)w;                           w += (long)S * S * 4;   // scores / probs
    float* attn  = (float*)w;                           w += SH * 4;
    float* haft  = (float*)w;                           w += SH * 4;
    float* ln2   = (float*)w;                           w += SH * 4;
    float* gates = (float*)w;                           w += (long)S * E * 4;
    float* fc1   = (float*)w;                           w += (long)S * F * 4;
    float* ybuf  = (float*)w;                           w += SH * 4;
    float* macc  = (float*)w;                           w += SH * 4;
    float* projo = sc;   // reuse scores buffer after attention

    auto gemm_tn = [&](const float* A, const float* B, float* C,
                       int M, int N, int K, int lda, int ldb, int ldc,
                       float alpha, int causal) {
        dim3 grid((N + TN - 1) / TN, (M + TM - 1) / TM);
        gemm_bf16_wmma<0><<<grid, 256, 0, stream>>>(A, B, C, M, N, K,
                                                    lda, ldb, ldc, alpha, causal);
    };
    auto gemm_nn = [&](const float* A, const float* B, float* C,
                       int M, int N, int K, int lda, int ldb, int ldc,
                       float alpha) {
        dim3 grid((N + TN - 1) / TN, (M + TM - 1) / TM);
        gemm_bf16_wmma<1><<<grid, 256, 0, stream>>>(A, B, C, M, N, K,
                                                    lda, ldb, ldc, alpha, 0);
    };

    const int EW_BLOCKS = 4096;

    // 1) LN1
    layernorm_kernel<<<S, 256, 0, stream>>>(hidden, ln1_w, ln1_b, ln1, H);

    // 2) QKV = ln1 @ qkv_w^T     [S x 1536]
    gemm_tn(ln1, qkv_w, qkv, S, QKVO, H, H, H, QKVO, 1.0f, 0);

    // 3) Attention, head by head (GQA: kv = hq/4)
    for (int hq = 0; hq < Hq; ++hq) {
        int kv = hq / 4, j = hq % 4;
        const float* Q = qkv + kv * 384 + j * D;        // lda = 1536
        const float* K = qkv + kv * 384 + 256;          // ldb = 1536
        const float* V = qkv + kv * 384 + 320;          // ldb = 1536

        // scores = Q @ K^T * 1/sqrt(D), causal blocks skipped
        gemm_tn(Q, K, sc, S, S, D, QKVO, QKVO, S, 0.125f, 1);
        softmax_causal_kernel<<<S, 256, 0, stream>>>(sc, S);
        // attn[:, hq*64 : hq*64+64] = probs @ V
        gemm_nn(sc, V, attn + hq * D, S, D, S, S, QKVO, H, 1.0f);
    }

    // 4) proj + residual
    gemm_tn(attn, proj_w, projo, S, H, H, H, H, H, 1.0f, 0);
    add_kernel<<<EW_BLOCKS, 256, 0, stream>>>(hidden, projo, haft, SH);

    // 5) LN2 + router
    layernorm_kernel<<<S, 256, 0, stream>>>(haft, ln2_w, ln2_b, ln2, H);
    router_kernel<<<S, 256, 0, stream>>>(ln2, rtr_w, gates, H, E);

    // 6) MoE: dense per-expert GEMMs, gated accumulate (gates are zero
    //    except the renormalized top-2, matching the reference math)
    zero_kernel<<<EW_BLOCKS, 256, 0, stream>>>(macc, SH);
    for (int e = 0; e < E; ++e) {
        const float* w1 = moe_w1 + (long)e * H * F;     // [H][F]
        const float* w2 = moe_w2 + (long)e * F * H;     // [F][H]
        gemm_nn(ln2, w1, fc1, S, F, H, H, F, F, 1.0f);
        gelu_kernel<<<EW_BLOCKS, 256, 0, stream>>>(fc1, (long)S * F);
        gemm_nn(fc1, w2, ybuf, S, H, F, F, H, H, 1.0f);
        scale_accum_kernel<<<EW_BLOCKS, 256, 0, stream>>>(ybuf, gates, macc, H, E, e, SH);
    }

    // 7) final residual
    add_kernel<<<EW_BLOCKS, 256, 0, stream>>>(haft, macc, out, SH);
}
